// AttnBlock_32701880991984
// MI455X (gfx1250) — compile-verified, matrix operand order checked by reading, not run on hardware
//
#include <hip/hip_runtime.h>
#include <hip/hip_bf16.h>

#define B_ 16
#define C_ 128
#define HW_ 4096
#define L_ 512
#define ND_ 256
#define GROUPS_ 32

typedef __attribute__((ext_vector_type(16))) _Float16 v16h;
typedef __attribute__((ext_vector_type(8)))  _Float16 v8h;
typedef __attribute__((ext_vector_type(8)))  float    v8f;

// ---------------------------------------------------------------------------
// WMMA helper: D = A(16x32 f16) * B(32x16 f16) + C(16x16 f32)
// ---------------------------------------------------------------------------
__device__ __forceinline__ v8f wmma_f16(v16h a, v16h b, v8f c) {
  return __builtin_amdgcn_wmma_f32_16x16x32_f16(
      /*neg_a=*/false, a, /*neg_b=*/false, b,
      /*c_mod=*/(short)0, c, /*reuse_a=*/false, /*reuse_b=*/false);
}

// A-operand loader from row-major f16 [rows x ld] at (row0, k0).
// ISA layout: lanes 0-15 -> rows 0-15 with K 0..7 (v0-3) and 16..23 (v4-7);
// lanes 16-31 -> same rows with K 8..15 and 24..31.
// B-operand uses the same loader on the *transposed* matrix (col per lane).
__device__ __forceinline__ v16h load_f16_tile(const _Float16* __restrict__ p,
                                              int ld, int row0, int k0, int lane) {
  const int r  = lane & 15;
  const int kb = (lane >> 4) << 3;
  const _Float16* base = p + (size_t)(row0 + r) * ld + k0 + kb;
  v8h lo = *reinterpret_cast<const v8h*>(base);
  v8h hi = *reinterpret_cast<const v8h*>(base + 16);
  v16h out;
#pragma unroll
  for (int i = 0; i < 8; ++i) { out[i] = lo[i]; out[i + 8] = hi[i]; }
  return out;
}

// A-operand loader from fp32 LDS [16 x pitch] with f32->f16 convert.
__device__ __forceinline__ v16h load_f32_tile_cvt(const float* __restrict__ s,
                                                  int pitch, int k0, int lane) {
  const int r  = lane & 15;
  const int kb = (lane >> 4) << 3;
  const float* base = s + r * pitch + k0 + kb;
  v16h out;
#pragma unroll
  for (int i = 0; i < 8; ++i) {
    out[i]     = (_Float16)base[i];
    out[i + 8] = (_Float16)base[i + 16];
  }
  return out;
}

// ---------------------------------------------------------------------------
// Kernel 1: convert fp32 weights to f16
// ---------------------------------------------------------------------------
__global__ void k_cvt(const float* __restrict__ qw, const float* __restrict__ pw,
                      const float* __restrict__ ndw,
                      _Float16* __restrict__ qw16, _Float16* __restrict__ pw16,
                      _Float16* __restrict__ ndw16) {
  int i = blockIdx.x * 256 + threadIdx.x;
  if (i < C_ * C_) {
    qw16[i] = (_Float16)qw[i];
    pw16[i] = (_Float16)pw[i];
  }
  if (i < C_ * ND_) ndw16[i] = (_Float16)ndw[i];
}

// ---------------------------------------------------------------------------
// Kernel 2: GroupNorm; writes hn transposed as f16 [b][n][c]
// ---------------------------------------------------------------------------
__global__ __launch_bounds__(256) void k_gnorm(const float* __restrict__ x,
                                               const float* __restrict__ gamma,
                                               const float* __restrict__ beta,
                                               _Float16* __restrict__ hn16) {
  __shared__ float rs[256], rs2[256];
  const int b = blockIdx.x, g = blockIdx.y;
  const int CPG = C_ / GROUPS_;                       // 4
  const float* xp = x + ((size_t)b * C_ + g * CPG) * HW_;
  float s = 0.f, s2 = 0.f;
  for (int i = threadIdx.x; i < CPG * HW_; i += 256) {
    float v = xp[i]; s += v; s2 += v * v;
  }
  rs[threadIdx.x] = s; rs2[threadIdx.x] = s2;
  __syncthreads();
  for (int off = 128; off > 0; off >>= 1) {
    if (threadIdx.x < off) {
      rs[threadIdx.x]  += rs[threadIdx.x + off];
      rs2[threadIdx.x] += rs2[threadIdx.x + off];
    }
    __syncthreads();
  }
  const float inv_n = 1.0f / (float)(CPG * HW_);
  const float mean = rs[0] * inv_n;
  const float var  = rs2[0] * inv_n - mean * mean;
  const float rstd = rsqrtf(var + 1e-6f);
  _Float16* dst = hn16 + (size_t)b * HW_ * C_;
  for (int i = threadIdx.x; i < CPG * HW_; i += 256) {
    int cg = i >> 12;            // i / 4096
    int n  = i & (HW_ - 1);
    int c  = g * CPG + cg;
    float v = (xp[i] - mean) * rstd * gamma[c] + beta[c];
    dst[(size_t)n * C_ + c] = (_Float16)v;
  }
}

// ---------------------------------------------------------------------------
// Kernel 3: Q = hn_t[4096x128] @ q_w^T[128x128] + q_b -> f16 [b][n][c]
// One 16-row x 128-col strip per wave: A tiles preloaded once, B looped.
// ---------------------------------------------------------------------------
__global__ __launch_bounds__(256) void k_qproj(const _Float16* __restrict__ hn16,
                                               const _Float16* __restrict__ qw16,
                                               const float* __restrict__ q_b,
                                               _Float16* __restrict__ q16) {
  const int b = blockIdx.x;
  const int lane = threadIdx.x & 31;
  const int tn = blockIdx.y * 8 + (threadIdx.x >> 5);   // 0..255 row strips
  const _Float16* A = hn16 + (size_t)b * HW_ * C_;
  _Float16* Q = q16 + (size_t)b * HW_ * C_;

  v16h aT[4];
#pragma unroll
  for (int kk = 0; kk < 4; ++kk)
    aT[kk] = load_f16_tile(A, C_, tn * 16, kk * 32, lane);

  const int rbase = tn * 16 + ((lane >> 4) << 3);
#pragma unroll
  for (int tc = 0; tc < 8; ++tc) {
    v8f acc = {};
#pragma unroll
    for (int kk = 0; kk < 4; ++kk) {
      v16h bm = load_f16_tile(qw16, C_, tc * 16, kk * 32, lane);  // q_w is [o][c]
      acc = wmma_f16(aT[kk], bm, acc);
    }
    const int col = tc * 16 + (lane & 15);
    const float bias = q_b[col];
#pragma unroll
    for (int r = 0; r < 8; ++r)
      Q[(size_t)(rbase + r) * C_ + col] = (_Float16)(acc[r] + bias);
  }
}

// ---------------------------------------------------------------------------
// Kernel 4: KV = silu(nd[512x256]) @ nd_w^T[256x128] + nd_b
// SiLU computed once per element into register-resident f16 A tiles.
// Writes f16 [l][c] and transposed [c][l].
// ---------------------------------------------------------------------------
__global__ __launch_bounds__(256) void k_kv(const float* __restrict__ nd,
                                            const _Float16* __restrict__ ndw16,
                                            const float* __restrict__ nd_b,
                                            _Float16* __restrict__ kv16,
                                            _Float16* __restrict__ kvT16) {
  const int b = blockIdx.x;
  const int lane = threadIdx.x & 31;
  const int tl = blockIdx.y * 8 + (threadIdx.x >> 5);   // 0..31 row strips
  const float* A = nd + (size_t)b * L_ * ND_;
  const int r = lane & 15, kb = (lane >> 4) << 3;

  v16h aT[8];
#pragma unroll
  for (int kk = 0; kk < 8; ++kk) {
    const float* ap = A + (size_t)(tl * 16 + r) * ND_ + kk * 32 + kb;
#pragma unroll
    for (int j = 0; j < 8; ++j) {
      float v0 = ap[j];
      float v1 = ap[j + 16];
      aT[kk][j]     = (_Float16)(v0 / (1.0f + __expf(-v0)));   // silu
      aT[kk][j + 8] = (_Float16)(v1 / (1.0f + __expf(-v1)));
    }
  }

  _Float16* kvb = kv16 + (size_t)b * L_ * C_;
  _Float16* kvt = kvT16 + (size_t)b * C_ * L_;
  const int rbase = tl * 16 + ((lane >> 4) << 3);
#pragma unroll
  for (int tc = 0; tc < 8; ++tc) {
    v8f acc = {};
#pragma unroll
    for (int kk = 0; kk < 8; ++kk) {
      v16h bm = load_f16_tile(ndw16, ND_, tc * 16, kk * 32, lane); // nd_w is [c][nd]
      acc = wmma_f16(aT[kk], bm, acc);
    }
    const int col = tc * 16 + (lane & 15);
    const float bias = nd_b[col];
#pragma unroll
    for (int rr = 0; rr < 8; ++rr) {
      float v = acc[rr] + bias;
      kvb[(size_t)(rbase + rr) * C_ + col] = (_Float16)v;
      kvt[(size_t)col * L_ + rbase + rr]   = (_Float16)v;
    }
  }
}

// ---------------------------------------------------------------------------
// Kernel 5: fused attention + proj + residual.
// Block = 2 waves; each wave owns 16 query rows. L=512 fits entirely, so
// softmax is exact (no online rescale). 64 KB static LDS.
// ---------------------------------------------------------------------------
__global__ __launch_bounds__(64) void k_attn(const _Float16* __restrict__ q16,
                                             const _Float16* __restrict__ kv16,
                                             const _Float16* __restrict__ kvT16,
                                             const _Float16* __restrict__ projw16,
                                             const float* __restrict__ proj_b,
                                             const float* __restrict__ x,
                                             float* __restrict__ out) {
  __shared__ float sS[2][16][L_];            // 64 KB: per-wave 16x512 logits
  const int b = blockIdx.x;
  const int wave = threadIdx.x >> 5;
  const int lane = threadIdx.x & 31;
  const int m0 = blockIdx.y * 32 + wave * 16;          // query-row base
  const float scale = 0.088388347648318447f;           // 128^-0.5

  const _Float16* qb   = q16   + (size_t)b * HW_ * C_;
  const _Float16* kvb  = kv16  + (size_t)b * L_ * C_;
  const _Float16* kvtb = kvT16 + (size_t)b * C_ * L_;
  float (*S)[L_] = sS[wave];

  // ---- S = Q @ KV^T (32 column tiles, K=128); Q tiles register-resident ----
  v16h aQ[4];
#pragma unroll
  for (int kk = 0; kk < 4; ++kk)
    aQ[kk] = load_f16_tile(qb, C_, m0, kk * 32, lane);

  for (int nt = 0; nt < 32; ++nt) {
    v8f acc = {};
#pragma unroll
    for (int kk = 0; kk < 4; ++kk) {
      v16h bm = load_f16_tile(kvb, C_, nt * 16, kk * 32, lane); // KV row = col of KV^T
      acc = wmma_f16(aQ[kk], bm, acc);
    }
    const int col = nt * 16 + (lane & 15);
    const int rbase = (lane >> 4) << 3;
#pragma unroll
    for (int r = 0; r < 8; ++r) S[rbase + r][col] = acc[r] * scale;
  }
  __syncthreads();

  // ---- exact softmax over L: 2 lanes per row (lane, lane^16) ----
  {
    const int row = lane & 15;
    const int k0 = (lane >> 4) * (L_ / 2);
    float* srow = S[row];
    float mx = -3.0e38f;
    for (int k = k0; k < k0 + L_ / 2; ++k) mx = fmaxf(mx, srow[k]);
    mx = fmaxf(mx, __shfl_xor(mx, 16, 32));
    float sum = 0.f;
    for (int k = k0; k < k0 + L_ / 2; ++k) {
      float e = __expf(srow[k] - mx);
      srow[k] = e;
      sum += e;
    }
    sum += __shfl_xor(sum, 16, 32);
    const float inv = 1.0f / sum;
    for (int k = k0; k < k0 + L_ / 2; ++k) srow[k] *= inv;
  }
  __syncthreads();

  // ---- O = P @ KV (8 channel tiles, K=512); A from LDS (f32->f16) ----
  v8f oacc[8];
#pragma unroll
  for (int i = 0; i < 8; ++i) oacc[i] = (v8f){};
  for (int k0 = 0; k0 < L_; k0 += 32) {
    v16h a = load_f32_tile_cvt(&S[0][0], L_, k0, lane);
#pragma unroll
    for (int ct = 0; ct < 8; ++ct) {
      v16h bm = load_f16_tile(kvtb, L_, ct * 16, k0, lane); // kvT row = col of KV
      oacc[ct] = wmma_f16(a, bm, oacc[ct]);
    }
  }
  __syncthreads();

  // ---- LDS transpose: D-layout -> A-layout as f16 [16][128] ----
  _Float16* sO = reinterpret_cast<_Float16*>(&S[0][0]);
  {
    const int rbase = (lane >> 4) << 3;
#pragma unroll
    for (int ct = 0; ct < 8; ++ct) {
      const int col = ct * 16 + (lane & 15);
#pragma unroll
      for (int r = 0; r < 8; ++r)
        sO[(rbase + r) * C_ + col] = (_Float16)oacc[ct][r];
    }
  }
  __syncthreads();

  // ---- Y = O @ proj_w^T + proj_b ; out = x + Y (O tiles register-resident) ----
  v16h aO[4];
#pragma unroll
  for (int kk = 0; kk < 4; ++kk)
    aO[kk] = load_f16_tile(sO, C_, 0, kk * 32, lane);

#pragma unroll
  for (int ot = 0; ot < 8; ++ot) {
    v8f acc = {};
#pragma unroll
    for (int kk = 0; kk < 4; ++kk) {
      v16h bm = load_f16_tile(projw16, C_, ot * 16, kk * 32, lane);
      acc = wmma_f16(aO[kk], bm, acc);
    }
    const int o = ot * 16 + (lane & 15);
    const int rbase = m0 + ((lane >> 4) << 3);
    const float bias = proj_b[o];
    const size_t base = ((size_t)b * C_ + o) * HW_;
#pragma unroll
    for (int r = 0; r < 8; ++r) {
      const size_t idx = base + rbase + r;
      out[idx] = x[idx] + acc[r] + bias;
    }
  }
}

// ---------------------------------------------------------------------------
// Host launch
// ---------------------------------------------------------------------------
extern "C" void kernel_launch(void* const* d_in, const int* in_sizes, int n_in,
                              void* d_out, int out_size, void* d_ws, size_t ws_size,
                              hipStream_t stream) {
  const float* x   = (const float*)d_in[0];
  const float* nd  = (const float*)d_in[1];
  const float* gng = (const float*)d_in[2];
  const float* gnb = (const float*)d_in[3];
  const float* qw  = (const float*)d_in[4];
  const float* qb  = (const float*)d_in[5];
  const float* ndw = (const float*)d_in[6];
  const float* ndb = (const float*)d_in[7];
  const float* pw  = (const float*)d_in[8];
  const float* pb  = (const float*)d_in[9];
  float* out = (float*)d_out;

  char* ws = (char*)d_ws;
  _Float16* hn16  = (_Float16*)(ws);                               // 16 MiB
  _Float16* q16   = (_Float16*)(ws + ((size_t)16 << 20));          // 16 MiB
  _Float16* kv16  = (_Float16*)(ws + ((size_t)32 << 20));          // 2 MiB
  _Float16* kvT16 = (_Float16*)(ws + ((size_t)34 << 20));          // 2 MiB
  _Float16* qw16  = (_Float16*)(ws + ((size_t)36 << 20));          // 32 KiB
  _Float16* pw16  = (_Float16*)(ws + ((size_t)36 << 20) + 32768);  // 32 KiB
  _Float16* ndw16 = (_Float16*)(ws + ((size_t)36 << 20) + 65536);  // 64 KiB

  k_cvt<<<dim3((C_ * ND_ + 255) / 256), dim3(256), 0, stream>>>(
      qw, pw, ndw, qw16, pw16, ndw16);
  k_gnorm<<<dim3(B_, GROUPS_), dim3(256), 0, stream>>>(x, gng, gnb, hn16);
  k_qproj<<<dim3(B_, 32), dim3(256), 0, stream>>>(hn16, qw16, qb, q16);
  k_kv<<<dim3(B_, 4), dim3(256), 0, stream>>>(nd, ndw16, ndb, kv16, kvT16);
  k_attn<<<dim3(B_, HW_ / 32), dim3(64), 0, stream>>>(
      q16, kv16, kvT16, pw16, pb, x, out);
}
